// SelectiveRecurrentFFN_69123203662001
// MI455X (gfx1250) — matrix operand order repaired and checked by reading, not compiled, not run on hardware
//
#include <hip/hip_runtime.h>

// ---------------------------------------------------------------------------
// SelectiveRecurrentFFN for MI455X (gfx1250, wave32, WMMA, LDS-staged weights,
// scheduler-directed WMMA/DS software pipelining via sched_group_barrier)
// B=4, S=2048, H=1024, N=1024  -> M = B*S = 8192 tokens
// ---------------------------------------------------------------------------

typedef unsigned short u16;
typedef unsigned int   u32;
typedef __attribute__((ext_vector_type(16))) __bf16 v16bf;
typedef __attribute__((ext_vector_type(8)))  float  v8f;
typedef __attribute__((ext_vector_type(4)))  u32    u32x4;

#define MTOK   8192
#define HDIM   1024
#define NDIM   1024
#define BATCH  4
#define SEQ    2048
#define CHUNK  256
#define NCHUNK (SEQ / CHUNK)   // 8
#define GW     40              // padded LDS row stride (u16): 80B, 16B-aligned rows

#define SGB(mask, size) __builtin_amdgcn_sched_group_barrier((mask), (size), 0)
// masks: 0x008 = MFMA/WMMA, 0x020 = VMEM read, 0x100 = DS read

// ---------------- helpers ----------------

__device__ __forceinline__ u16 f2bf(float x) {
    u32 u = __float_as_uint(x);
    u += 0x7FFFu + ((u >> 16) & 1u);       // round-to-nearest-even
    return (u16)(u >> 16);
}
__device__ __forceinline__ float sigmoidf_(float x) { return 1.0f / (1.0f + __expf(-x)); }
__device__ __forceinline__ float siluf_(float x)    { return x * sigmoidf_(x); }

union FragU { u32x4 u[2]; v16bf v; };

// A-matrix fragment, 16x32 bf16 (row-major source, ld in elements).
__device__ __forceinline__ FragU load_a_raw(const u16* __restrict__ A, int ld,
                                            int row, int k0, int lane) {
    int hf = lane >> 4;
    const u16* p = A + (size_t)row * ld + k0 + 8 * hf;
    FragU f;
    f.u[0] = *(const u32x4*)(p);
    f.u[1] = *(const u32x4*)(p + 16);
    return f;
}

// B-matrix fragment, 32x16 bf16 read from an LDS weight slab [rows][GW].
__device__ __forceinline__ v16bf lds_b_frag(const u16* base, int t, int lane) {
    int hf = lane >> 4;
    const u16* p = base + (16 * t + (lane & 15)) * GW + 8 * hf;
    FragU f;
    f.u[0] = *(const u32x4*)(p);
    f.u[1] = *(const u32x4*)(p + 16);
    return f.v;
}

#define WMMA_BF16(a, b, c) \
    __builtin_amdgcn_wmma_f32_16x16x32_bf16(false, (a), false, (b), (short)0, (c), false, false)

// ---------------- pack fp32 -> bf16 ----------------

__global__ __launch_bounds__(256)
void pack_bf16(const float* __restrict__ src, u16* __restrict__ dst, int n) {
    int i = blockIdx.x * blockDim.x + threadIdx.x;
    int stride = gridDim.x * blockDim.x;
    for (; i < n; i += stride) dst[i] = f2bf(src[i]);
}

// ---------------- fused gate GEMM ----------------
// f = tanh(x Wf^T), wc = sigmoid(x Wi^T) * silu(x Wv^T)
// block = 256 threads (8 waves). block tile = 128 rows x 64 cols; wave tile 16x64.
// grid = (M/128, N/64) = (64, 16)
// NOTE: final-iteration prefetch reads run past the logical array but stay
// inside the contiguous workspace carve (values unused) -- removes control
// flow so prefetch + compute share one scheduling region.

__global__ __launch_bounds__(256)
void gates_gemm(const u16* __restrict__ xb,
                const u16* __restrict__ wfb,
                const u16* __restrict__ wib,
                const u16* __restrict__ wvb,
                float* __restrict__ fgate,
                float* __restrict__ wc) {
    __shared__ u16 wsh[2][3][64 * GW];            // 30 KB

    const int tid  = threadIdx.x;
    const int lane = tid & 31;
    const int wave = tid >> 5;
    const int m0 = blockIdx.x * 128 + wave * 16;
    const int n0 = blockIdx.y * 64;
    const int arow = m0 + (lane & 15);

    // cooperative staging slot: one b128 (8 u16) per gate per thread per chunk
    const int srow = tid >> 2;                    // 0..63
    const int skid = (tid & 3) * 8;               // 0,8,16,24
    const size_t goff = (size_t)(n0 + srow) * HDIM + skid;
    u16* const sdst_f = &wsh[0][0][srow * GW + skid];
    const int sbufstride = 3 * 64 * GW;

    v8f accf[4], acci[4], accv[4];
#pragma unroll
    for (int t = 0; t < 4; ++t) { accf[t] = (v8f)0.0f; acci[t] = (v8f)0.0f; accv[t] = (v8f)0.0f; }

    // ---- prologue: stage chunk 0, load A frag 0 ----
    {
        u32x4 rf = *(const u32x4*)(wfb + goff);
        u32x4 ri = *(const u32x4*)(wib + goff);
        u32x4 rv = *(const u32x4*)(wvb + goff);
        *(u32x4*)(sdst_f)               = rf;
        *(u32x4*)(sdst_f + 64 * GW)     = ri;
        *(u32x4*)(sdst_f + 2 * 64 * GW) = rv;
    }
    __syncthreads();
    FragU a_cur = load_a_raw(xb, HDIM, arow, 0, lane);

    for (int kc = 0; kc < 32; ++kc) {
        const int cur = kc & 1;
        const int k0n = (kc + 1) * 32;            // kc==31: OOB-but-mapped, unused
        // prefetch next chunk (A first: next iter's WMMAs can start at loadcnt<=3)
        FragU a_next = load_a_raw(xb, HDIM, arow, k0n, lane);
        u32x4 rf = *(const u32x4*)(wfb + goff + k0n);
        u32x4 ri = *(const u32x4*)(wib + goff + k0n);
        u32x4 rv = *(const u32x4*)(wvb + goff + k0n);

        // 12 WMMAs; item j: tile t = j/3, gate g = j%3
        const u16* bases[3] = { &wsh[cur][0][0], &wsh[cur][1][0], &wsh[cur][2][0] };
#pragma unroll
        for (int j = 0; j < 12; ++j) {
            const int t = j / 3, g = j % 3;
            v16bf bw = lds_b_frag(bases[g], t, lane);
            if      (g == 0) accf[t] = WMMA_BF16(a_cur.v, bw, accf[t]);
            else if (g == 1) acci[t] = WMMA_BF16(a_cur.v, bw, acci[t]);
            else             accv[t] = WMMA_BF16(a_cur.v, bw, accv[t]);
        }
        // scheduler pipeline: [DS x4][VMEM x5] ([WMMA][DS x2]) x10 [WMMA x2]
        SGB(0x100, 4);
        SGB(0x020, 5);
#pragma unroll
        for (int j = 0; j < 10; ++j) { SGB(0x008, 1); SGB(0x100, 2); }
        SGB(0x008, 2);

        __syncthreads();                          // all waves done reading buf cur^1
        u16* d = sdst_f + (cur ^ 1) * sbufstride;
        *(u32x4*)(d)               = rf;
        *(u32x4*)(d + 64 * GW)     = ri;
        *(u32x4*)(d + 2 * 64 * GW) = rv;
        __syncthreads();
        a_cur = a_next;
    }

    // C/D layout: VGPR r -> row m0 + 8*(lane>>4) + r, col n0 + 16*t + (lane&15)
    const int hf = lane >> 4;
    const int nlo = lane & 15;
#pragma unroll
    for (int t = 0; t < 4; ++t) {
#pragma unroll
        for (int r = 0; r < 8; ++r) {
            int m = m0 + 8 * hf + r;
            int n = n0 + 16 * t + nlo;
            size_t idx = (size_t)m * NDIM + n;
            fgate[idx] = tanhf(accf[t][r]);
            wc[idx]    = sigmoidf_(acci[t][r]) * siluf_(accv[t][r]);
        }
    }
}

// ---------------- chunked scan: state = f*state + wc ----------------

__global__ __launch_bounds__(256)
void scan_chunk(const float* __restrict__ fgate, const float* __restrict__ wc,
                float* __restrict__ R, float* __restrict__ P) {
    int tid = blockIdx.x * 256 + threadIdx.x;          // 0 .. 32767
    int n = tid & (NDIM - 1);
    int c = (tid >> 10) & (NCHUNK - 1);
    int b = tid >> 13;
    size_t base = ((size_t)(b * SEQ + c * CHUNK)) * NDIM + n;
    float st = 0.0f, pr = 1.0f;
#pragma unroll 4
    for (int i = 0; i < CHUNK; ++i) {
        float f = fgate[base + (size_t)i * NDIM];
        float w = wc[base + (size_t)i * NDIM];
        st = fmaf(f, st, w);
        pr *= f;
    }
    int o = (b * NCHUNK + c) * NDIM + n;
    R[o] = st;
    P[o] = pr;
}

__global__ __launch_bounds__(256)
void scan_carry(const float* __restrict__ R, const float* __restrict__ P,
                float* __restrict__ Si) {
    int tid = blockIdx.x * 256 + threadIdx.x;          // 0 .. 4095
    int n = tid & (NDIM - 1);
    int b = tid >> 10;
    float carry = 0.0f;
#pragma unroll
    for (int c = 0; c < NCHUNK; ++c) {
        int o = (b * NCHUNK + c) * NDIM + n;
        Si[o] = carry;
        carry = fmaf(P[o], carry, R[o]);
    }
}

__global__ __launch_bounds__(256)
void scan_apply(const float* __restrict__ fgate, const float* __restrict__ wc,
                const float* __restrict__ Si, u16* __restrict__ hs) {
    int tid = blockIdx.x * 256 + threadIdx.x;          // 0 .. 32767
    int n = tid & (NDIM - 1);
    int c = (tid >> 10) & (NCHUNK - 1);
    int b = tid >> 13;
    size_t base = ((size_t)(b * SEQ + c * CHUNK)) * NDIM + n;
    float st = Si[(b * NCHUNK + c) * NDIM + n];
#pragma unroll 4
    for (int i = 0; i < CHUNK; ++i) {
        size_t idx = base + (size_t)i * NDIM;
        st = fmaf(fgate[idx], st, wc[idx]);
        hs[idx] = f2bf(siluf_(st));
    }
}

// ---------------- output GEMM: out = hs . Wo^T  (Wo is [H, N] row-major) ------
// block = 256 threads (8 waves). block tile = 128 rows x 128 cols; wave tile 16x128.
// grid = (M/128, H/128) = (64, 8)

__global__ __launch_bounds__(256)
void out_gemm(const u16* __restrict__ hs, const u16* __restrict__ wob,
              float* __restrict__ out) {
    __shared__ u16 wsh[2][128 * GW];              // 20 KB

    const int tid  = threadIdx.x;
    const int lane = tid & 31;
    const int wave = tid >> 5;
    const int m0 = blockIdx.x * 128 + wave * 16;
    const int n0 = blockIdx.y * 128;
    const int arow = m0 + (lane & 15);

    // staging: 128 rows x 32 k = 512 b128 chunks; 2 per thread
    const int c0 = tid * 2;
    const int srow0 = c0 >> 2,       skid0 = (c0 & 3) * 8;
    const int srow1 = (c0 + 1) >> 2, skid1 = ((c0 + 1) & 3) * 8;
    const size_t goff0 = (size_t)(n0 + srow0) * NDIM + skid0;
    const size_t goff1 = (size_t)(n0 + srow1) * NDIM + skid1;
    u16* const sdst0 = &wsh[0][srow0 * GW + skid0];
    u16* const sdst1 = &wsh[0][srow1 * GW + skid1];
    const int sbufstride = 128 * GW;

    v8f acc[8];
#pragma unroll
    for (int t = 0; t < 8; ++t) acc[t] = (v8f)0.0f;

    {
        u32x4 r0 = *(const u32x4*)(wob + goff0);
        u32x4 r1 = *(const u32x4*)(wob + goff1);
        *(u32x4*)(sdst0) = r0;
        *(u32x4*)(sdst1) = r1;
    }
    __syncthreads();
    FragU a_cur = load_a_raw(hs, NDIM, arow, 0, lane);

    for (int kc = 0; kc < 32; ++kc) {
        const int cur = kc & 1;
        const int k0n = (kc + 1) * 32;            // kc==31: OOB-but-mapped, unused
        FragU a_next = load_a_raw(hs, NDIM, arow, k0n, lane);
        u32x4 r0 = *(const u32x4*)(wob + goff0 + k0n);
        u32x4 r1 = *(const u32x4*)(wob + goff1 + k0n);

        const u16* bbase = &wsh[cur][0];
#pragma unroll
        for (int t = 0; t < 8; ++t) {
            v16bf bw = lds_b_frag(bbase, t, lane);
            acc[t] = WMMA_BF16(a_cur.v, bw, acc[t]);
        }
        // scheduler pipeline: [DS x4][VMEM x4] ([WMMA][DS x2]) x6 [WMMA x2]
        SGB(0x100, 4);
        SGB(0x020, 4);
#pragma unroll
        for (int j = 0; j < 6; ++j) { SGB(0x008, 1); SGB(0x100, 2); }
        SGB(0x008, 2);

        __syncthreads();
        u16* d0 = sdst0 + (cur ^ 1) * sbufstride;
        u16* d1 = sdst1 + (cur ^ 1) * sbufstride;
        *(u32x4*)(d0) = r0;
        *(u32x4*)(d1) = r1;
        __syncthreads();
        a_cur = a_next;
    }

    const int hf = lane >> 4;
    const int nlo = lane & 15;
#pragma unroll
    for (int t = 0; t < 8; ++t) {
#pragma unroll
        for (int r = 0; r < 8; ++r) {
            int m = m0 + 8 * hf + r;
            int h = n0 + 16 * t + nlo;
            out[(size_t)m * HDIM + h] = acc[t][r];
        }
    }
}

// ---------------- host launcher ----------------

extern "C" void kernel_launch(void* const* d_in, const int* in_sizes, int n_in,
                              void* d_out, int out_size, void* d_ws, size_t ws_size,
                              hipStream_t stream) {
    const float* x  = (const float*)d_in[0];   // [B,S,H]
    const float* Wf = (const float*)d_in[1];   // [N,H]
    const float* Wi = (const float*)d_in[2];   // [N,H]
    const float* Wv = (const float*)d_in[3];   // [N,H]
    const float* Wo = (const float*)d_in[4];   // [H,N]
    float* out = (float*)d_out;                // [B,S,H]

    // workspace carve (~104 MB, contiguous -- OOB prefetch tails stay mapped)
    char* p = (char*)d_ws;
    u16* xb  = (u16*)p;  p += (size_t)MTOK * HDIM * sizeof(u16);
    u16* wfb = (u16*)p;  p += (size_t)NDIM * HDIM * sizeof(u16);
    u16* wib = (u16*)p;  p += (size_t)NDIM * HDIM * sizeof(u16);
    u16* wvb = (u16*)p;  p += (size_t)NDIM * HDIM * sizeof(u16);
    u16* wob = (u16*)p;  p += (size_t)HDIM * NDIM * sizeof(u16);
    float* fgate = (float*)p; p += (size_t)MTOK * NDIM * sizeof(float);
    float* wc    = (float*)p; p += (size_t)MTOK * NDIM * sizeof(float);
    u16*   hs    = (u16*)p;   p += (size_t)MTOK * NDIM * sizeof(u16);
    float* R  = (float*)p; p += (size_t)BATCH * NCHUNK * NDIM * sizeof(float);
    float* P  = (float*)p; p += (size_t)BATCH * NCHUNK * NDIM * sizeof(float);
    float* Si = (float*)p; p += (size_t)BATCH * NCHUNK * NDIM * sizeof(float);

    // 1) pack to bf16
    pack_bf16<<<4096, 256, 0, stream>>>(x,  xb,  MTOK * HDIM);
    pack_bf16<<<1024, 256, 0, stream>>>(Wf, wfb, NDIM * HDIM);
    pack_bf16<<<1024, 256, 0, stream>>>(Wi, wib, NDIM * HDIM);
    pack_bf16<<<1024, 256, 0, stream>>>(Wv, wvb, NDIM * HDIM);
    pack_bf16<<<1024, 256, 0, stream>>>(Wo, wob, HDIM * NDIM);

    // 2) fused gate GEMMs (tanh / sigmoid*silu epilogue)
    gates_gemm<<<dim3(MTOK / 128, NDIM / 64), 256, 0, stream>>>(xb, wfb, wib, wvb, fgate, wc);

    // 3) chunked diagonal scan
    scan_chunk<<<(BATCH * NCHUNK * NDIM) / 256, 256, 0, stream>>>(fgate, wc, R, P);
    scan_carry<<<(BATCH * NDIM) / 256, 256, 0, stream>>>(R, P, Si);
    scan_apply<<<(BATCH * NCHUNK * NDIM) / 256, 256, 0, stream>>>(fgate, wc, Si, hs);

    // 4) output projection
    out_gemm<<<dim3(MTOK / 128, HDIM / 128), 256, 0, stream>>>(hs, wob, out);
}
